// MutilHeadSelfAttention_38328288149823
// MI455X (gfx1250) — matrix operand order repaired and checked by reading, not compiled, not run on hardware
//
#include <hip/hip_runtime.h>

// MI455X (gfx1250) multi-head self-attention + pointwise-merge.
// Compute-bound (~43 GFLOP vs ~3us of HBM traffic at 23.3 TB/s) -> all
// matmuls via v_wmma_f32_16x16x32_bf16 (wave32 WMMA, f32 accumulate).
// Round 4: fix __exp2f -> __builtin_amdgcn_exp2f (bare v_exp_f32).
// 64-key flash blocks (16 WMMA per iter), base-2 softmax, float4 stores,
// 16x32 GEMM tiles with A-fragment reuse.

typedef __attribute__((ext_vector_type(16))) __bf16 bf16x16;
typedef __attribute__((ext_vector_type(8)))  __bf16 bf16x8;
typedef __attribute__((ext_vector_type(4)))  __bf16 bf16x4;
typedef __attribute__((ext_vector_type(8)))  float  f32x8;

#define BATCH 2
#define SEQ   2048
#define CH    1024
#define NH    16
#define HD    64          // head dim = CH/NH
#define DSTR  (NH * SEQ)  // stride along d in the reshaped (B,dim,H,N) view

__device__ __forceinline__ float fast_exp2(float x) {
  return __builtin_amdgcn_exp2f(x);   // v_exp_f32
}

// 16-bit A-matrix 16x32 element->K mapping (contiguous order):
// lanes 0-15 hold K {0..15}; lanes 16-31 hold K {16..31}
__device__ __forceinline__ int a_kmap(int lane, int e) {
  int kb = (lane & 16) ? 16 : 0;
  return kb + e;
}

// ---- DPP butterfly reductions over each 16-lane row (pure VALU) ----------
// xor-masks {1,2,7,15} generate the full group over 16 lanes:
//   0xB1 = quad_perm(1,0,3,2) (xor1), 0x4E = quad_perm(2,3,0,1) (xor2)
//   0x141 = row_half_mirror (xor7),   0x140 = row_mirror (xor15)
#define DPP_STEP(v, ctrl, op)                                              \
  {                                                                        \
    int _s = __builtin_amdgcn_update_dpp(                                  \
        __builtin_bit_cast(int, v), __builtin_bit_cast(int, v),            \
        (ctrl), 0xf, 0xf, true);                                           \
    v = op(v, __builtin_bit_cast(float, _s));                              \
  }

__device__ __forceinline__ float addf(float a, float b) { return a + b; }

__device__ __forceinline__ float rowmax16(float v) {
  DPP_STEP(v, 0xB1, fmaxf)
  DPP_STEP(v, 0x4E, fmaxf)
  DPP_STEP(v, 0x141, fmaxf)
  DPP_STEP(v, 0x140, fmaxf)
  return v;
}
__device__ __forceinline__ float rowsum16(float v) {
  DPP_STEP(v, 0xB1, addf)
  DPP_STEP(v, 0x4E, addf)
  DPP_STEP(v, 0x141, addf)
  DPP_STEP(v, 0x140, addf)
  return v;
}

// ---------------------------------------------------------------------------
// Kernel 1: flash attention per (b, h). Workgroup = 256 thr (8 wave32), each
// wave owns 16 query rows; block covers 128 rows. 64-key blocks staged in LDS.
// Softmax in base-2 space (log2e folded into Q scale).
// Writes x[b, c=d*NH+h, n] (f32) to workspace.
// ---------------------------------------------------------------------------
__global__ __launch_bounds__(256)
void attn_flash_kernel(const float* __restrict__ q,
                       const float* __restrict__ k,
                       const float* __restrict__ v,
                       float* __restrict__ xout) {
  // Row strides are multiples of 16B so contiguous fragment reads stay aligned.
  __shared__ __align__(16) __bf16 kTt[64][80];      // [m][d] : K tile transposed
  __shared__ __align__(16) __bf16 vT [64][80];      // [d][m] : V tile
  __shared__ __align__(16) __bf16 pT [8][16][80];   // per-wave P bounce [row][k]

  const int lane = threadIdx.x & 31;
  const int wave = threadIdx.x >> 5;
  const int tid  = threadIdx.x;

  int bid = blockIdx.x;
  const int h  = bid % NH;    bid /= NH;
  const int b  = bid % BATCH; bid /= BATCH;
  const int n0 = bid * 128 + wave * 16;             // this wave's 16 query rows

  const size_t base  = (size_t)b * SEQ * CH + (size_t)h * SEQ; // + d*DSTR + n
  // 1/sqrt(N) * log2(e): base-2 softmax, one v_exp_f32 per exponential
  const float  scale = 0.02209708691207961f * 1.44269504088896340736f;

  // Q A-fragments for the two 32-wide d halves (d = K dim of the WMMA).
  // Inherently a gather (stride DSTR along d) but loaded only once.
  bf16x16 aQ0, aQ1;
  {
    const int nrow = n0 + (lane & 15);
#pragma unroll
    for (int e = 0; e < 16; ++e) {
      int d = a_kmap(lane, e);
      aQ0[e] = (__bf16)(q[base + (size_t)d * DSTR + nrow] * scale);
      aQ1[e] = (__bf16)(q[base + (size_t)(d + 32) * DSTR + nrow] * scale);
    }
  }

  f32x8 oAcc[4];
#pragma unroll
  for (int t = 0; t < 4; ++t) oAcc[t] = (f32x8){};
  float mrow[8], lrow[8];
#pragma unroll
  for (int r = 0; r < 8; ++r) { mrow[r] = -1e30f; lrow[r] = 0.0f; }

  const int koff = (lane & 16) ? 16 : 0;            // B-fragment K start
  const int kbA  = (lane & 16) ? 16 : 0;            // A-fragment K start

  for (int m0 = 0; m0 < SEQ; m0 += 64) {
    __syncthreads();
    // cooperative stage of K/V 64x64 f32 tiles -> LDS bf16, float4 loads
#pragma unroll
    for (int i = 0; i < 4; ++i) {
      int c = (i * 256 + tid) * 4;                  // 4096 floats per tile
      int d = c >> 6;
      int m = c & 63;
      const float4 kf = *(const float4*)&k[base + (size_t)d * DSTR + m0 + m];
      const float4 vf = *(const float4*)&v[base + (size_t)d * DSTR + m0 + m];
      // V tile: packed b64 store, contiguous along m
      bf16x4 vp = { (__bf16)vf.x, (__bf16)vf.y, (__bf16)vf.z, (__bf16)vf.w };
      *(bf16x4*)&vT[d][m] = vp;
      // K tile transposed: scatter across 4 rows
      kTt[m + 0][d] = (__bf16)kf.x;
      kTt[m + 1][d] = (__bf16)kf.y;
      kTt[m + 2][d] = (__bf16)kf.z;
      kTt[m + 3][d] = (__bf16)kf.w;
    }
    __syncthreads();

    // S tile (16 rows x 64 keys) = Q^T K : 4 col-blocks x 2 chained WMMAs
    f32x8 s[4];
#pragma unroll
    for (int cb = 0; cb < 4; ++cb) {
      const int mcol = cb * 16 + (lane & 15);
      bf16x16 b0 = *(const bf16x16*)&kTt[mcol][koff];       // d = 0..31
      bf16x16 b1 = *(const bf16x16*)&kTt[mcol][32 + koff];  // d = 32..63
      f32x8 c = (f32x8){};
      c = __builtin_amdgcn_wmma_f32_16x16x32_bf16(false, aQ0, false, b0,
                                                  (short)0, c, false, false);
      c = __builtin_amdgcn_wmma_f32_16x16x32_bf16(false, aQ1, false, b1,
                                                  (short)0, c, false, false);
      s[cb] = c;
    }

    // online softmax (base 2): element r of C/D tile is row (r + 8*(lane>=16))
#pragma unroll
    for (int r = 0; r < 8; ++r) {
      float smax  = rowmax16(fmaxf(fmaxf(s[0][r], s[1][r]),
                                   fmaxf(s[2][r], s[3][r])));
      float newm  = fmaxf(mrow[r], smax);
      float alpha = fast_exp2(mrow[r] - newm);
      float p0 = fast_exp2(s[0][r] - newm);
      float p1 = fast_exp2(s[1][r] - newm);
      float p2 = fast_exp2(s[2][r] - newm);
      float p3 = fast_exp2(s[3][r] - newm);
      lrow[r] = lrow[r] * alpha + rowsum16((p0 + p1) + (p2 + p3));
      mrow[r] = newm;
      int prow = r + ((lane >> 4) << 3);
      pT[wave][prow][lane & 15]        = (__bf16)p0;
      pT[wave][prow][16 + (lane & 15)] = (__bf16)p1;
      pT[wave][prow][32 + (lane & 15)] = (__bf16)p2;
      pT[wave][prow][48 + (lane & 15)] = (__bf16)p3;
#pragma unroll
      for (int t = 0; t < 4; ++t) oAcc[t][r] *= alpha;
    }
    asm volatile("s_wait_dscnt 0x0" ::: "memory"); // LDS P transpose fence

    // P back out of LDS in A-matrix layout: contiguous 8-element chunks
    bf16x16 aP0, aP1;                               // keys 0..31, 32..63
    {
      const int prow = lane & 15;
      bf16x8 l0 = *(const bf16x8*)&pT[wave][prow][kbA];
      bf16x8 h0 = *(const bf16x8*)&pT[wave][prow][kbA + 8];
      bf16x8 l1 = *(const bf16x8*)&pT[wave][prow][kbA + 32];
      bf16x8 h1 = *(const bf16x8*)&pT[wave][prow][kbA + 40];
      aP0 = __builtin_shufflevector(l0, h0, 0, 1, 2, 3, 4, 5, 6, 7,
                                    8, 9, 10, 11, 12, 13, 14, 15);
      aP1 = __builtin_shufflevector(l1, h1, 0, 1, 2, 3, 4, 5, 6, 7,
                                    8, 9, 10, 11, 12, 13, 14, 15);
    }
    // O(16x64) += P(16x64) * V^T(64x16) per 16-wide d block, 2 chained WMMAs
#pragma unroll
    for (int t = 0; t < 4; ++t) {
      const int d = t * 16 + (lane & 15);
      bf16x16 bV0 = *(const bf16x16*)&vT[d][koff];
      bf16x16 bV1 = *(const bf16x16*)&vT[d][32 + koff];
      oAcc[t] = __builtin_amdgcn_wmma_f32_16x16x32_bf16(false, aP0, false, bV0,
                                                        (short)0, oAcc[t],
                                                        false, false);
      oAcc[t] = __builtin_amdgcn_wmma_f32_16x16x32_bf16(false, aP1, false, bV1,
                                                        (short)0, oAcc[t],
                                                        false, false);
    }
  }

  // normalize and write x[b, d*NH + h, n] : rows contiguous -> float4 stores
  float inv[8];
#pragma unroll
  for (int r = 0; r < 8; ++r) inv[r] = 1.0f / lrow[r];
  const int nbase = n0 + ((lane >> 4) << 3);
#pragma unroll
  for (int t = 0; t < 4; ++t) {
    int d = t * 16 + (lane & 15);
    float* dst = &xout[(size_t)b * CH * SEQ + (size_t)(d * NH + h) * SEQ + nbase];
    float4 lo = { oAcc[t][0] * inv[0], oAcc[t][1] * inv[1],
                  oAcc[t][2] * inv[2], oAcc[t][3] * inv[3] };
    float4 hi = { oAcc[t][4] * inv[4], oAcc[t][5] * inv[5],
                  oAcc[t][6] * inv[6], oAcc[t][7] * inv[7] };
    *(float4*)dst       = lo;
    *(float4*)(dst + 4) = hi;
  }
}

// ---------------------------------------------------------------------------
// Kernel 2: out[b] = W(CHxCH) * X(CHxSEQ) + bias. Block = 8 waves covering a
// 16(o) x 256(n) tile; each wave does a 16x32 tile (A fragment reused across
// two WMMAs), K-loop of 32 steps.
// ---------------------------------------------------------------------------
__global__ __launch_bounds__(256)
void merge_gemm_kernel(const float* __restrict__ W,
                       const float* __restrict__ bias,
                       const float* __restrict__ X,
                       float* __restrict__ out) {
  const int lane = threadIdx.x & 31;
  const int wave = threadIdx.x >> 5;
  int bid = blockIdx.x;
  const int nblk = bid % (SEQ / 256); bid /= (SEQ / 256);
  const int oblk = bid % (CH / 16);   bid /= (CH / 16);
  const int b    = bid;
  const int o0 = oblk * 16;
  const int n0 = nblk * 256 + wave * 32;

  const float* Xb = X + (size_t)b * CH * SEQ;
  f32x8 acc0 = (f32x8){};
  f32x8 acc1 = (f32x8){};

  const int kbA   = (lane & 16) ? 16 : 0;
  const int koff  = (lane & 16) ? 16 : 0;
  const int orow  = o0 + (lane & 15);
  const int ncol0 = n0 + (lane & 15);
  const int ncol1 = ncol0 + 16;
  const float* Wrow = &W[(size_t)orow * CH];

  for (int k0 = 0; k0 < CH; k0 += 32) {
    // A fragment: one contiguous 16-float run of W's row -> float4 loads
    bf16x16 aW;
    {
      const float4 w0 = *(const float4*)&Wrow[k0 + kbA];
      const float4 w1 = *(const float4*)&Wrow[k0 + kbA + 4];
      const float4 w2 = *(const float4*)&Wrow[k0 + kbA + 8];
      const float4 w3 = *(const float4*)&Wrow[k0 + kbA + 12];
      aW[0]  = (__bf16)w0.x; aW[1]  = (__bf16)w0.y;
      aW[2]  = (__bf16)w0.z; aW[3]  = (__bf16)w0.w;
      aW[4]  = (__bf16)w1.x; aW[5]  = (__bf16)w1.y;
      aW[6]  = (__bf16)w1.z; aW[7]  = (__bf16)w1.w;
      aW[8]  = (__bf16)w2.x; aW[9]  = (__bf16)w2.y;
      aW[10] = (__bf16)w2.z; aW[11] = (__bf16)w2.w;
      aW[12] = (__bf16)w3.x; aW[13] = (__bf16)w3.y;
      aW[14] = (__bf16)w3.z; aW[15] = (__bf16)w3.w;
    }
    // B fragments: X[k, n] column strided (coalesced across lanes)
    bf16x16 bX0, bX1;
#pragma unroll
    for (int e = 0; e < 16; ++e) {
      const float* xr = &Xb[(size_t)(k0 + koff + e) * SEQ];
      bX0[e] = (__bf16)xr[ncol0];
      bX1[e] = (__bf16)xr[ncol1];
    }
    acc0 = __builtin_amdgcn_wmma_f32_16x16x32_bf16(false, aW, false, bX0,
                                                   (short)0, acc0, false, false);
    acc1 = __builtin_amdgcn_wmma_f32_16x16x32_bf16(false, aW, false, bX1,
                                                   (short)0, acc1, false, false);
  }

#pragma unroll
  for (int r = 0; r < 8; ++r) {
    int o = o0 + r + ((lane >> 4) << 3);
    float bo = bias[o];
    float* orow_p = &out[(size_t)b * CH * SEQ + (size_t)o * SEQ];
    orow_p[ncol0] = acc0[r] + bo;
    orow_p[ncol1] = acc1[r] + bo;
  }
}

extern "C" void kernel_launch(void* const* d_in, const int* in_sizes, int n_in,
                              void* d_out, int out_size, void* d_ws, size_t ws_size,
                              hipStream_t stream) {
  const float* q    = (const float*)d_in[0];
  const float* k    = (const float*)d_in[1];
  const float* v    = (const float*)d_in[2];
  const float* W    = (const float*)d_in[3];
  const float* bias = (const float*)d_in[4];
  float* out  = (float*)d_out;
  float* xbuf = (float*)d_ws;  // B*CH*SEQ floats = 16 MB intermediate

  (void)in_sizes; (void)n_in; (void)out_size; (void)ws_size;

  dim3 blk(256);
  dim3 g1(BATCH * NH * (SEQ / 128));          // 512 workgroups
  attn_flash_kernel<<<g1, blk, 0, stream>>>(q, k, v, xbuf);

  dim3 g2(BATCH * (CH / 16) * (SEQ / 256));   // 1024 workgroups
  merge_gemm_kernel<<<g2, blk, 0, stream>>>(W, bias, xbuf, out);
}
